// mLSTMInnerLayer_47880295416594
// MI455X (gfx1250) — matrix operand order repaired and checked by reading, not compiled, not run on hardware
//
#include <hip/hip_runtime.h>
#include <hip/hip_bf16.h>

typedef _Float16 v16h __attribute__((ext_vector_type(16)));
typedef float    v8f  __attribute__((ext_vector_type(8)));

#define BB   4
#define SS   2048
#define FF   1024
#define NHH  4
#define DHH  256     // FF/NHH
#define PP   256     // FF/BLK
#define BLKK 4
#define KCONV 4
#define TILES (SS/16)  // 128 row tiles per (b,h)

__device__ __forceinline__ float silu_f(float x) {
    return x / (1.f + __expf(-x));
}

// ---------------------------------------------------------------------------
// K1: causal depthwise conv + SiLU + block-diagonal q/k/v + gate logits.
// v is stored TRANSPOSED (b,h,dh,s) so K3's PV B-operand loads are contiguous.
// grid = B*S blocks, 256 threads (thread p owns 4-feature block p)
// ---------------------------------------------------------------------------
__global__ __launch_bounds__(256)
void k1_qkv_gates(const float* __restrict__ x_inner,
                  const float* __restrict__ conv_w, const float* __restrict__ conv_b,
                  const float* __restrict__ q_w, const float* __restrict__ k_w,
                  const float* __restrict__ v_w,
                  const float* __restrict__ ig_w, const float* __restrict__ ig_b,
                  const float* __restrict__ fg_w, const float* __restrict__ fg_b,
                  _Float16* __restrict__ q16, _Float16* __restrict__ k16,
                  _Float16* __restrict__ vT16, float* __restrict__ x_act,
                  float* __restrict__ ig_out, float* __restrict__ fgl_out)
{
    const int bs = blockIdx.x;
    const int b  = bs / SS, s = bs % SS;
    const int p  = threadIdx.x;
    const int f0 = p * BLKK;

    // causal window
    float xm[KCONV][BLKK];
#pragma unroll
    for (int kk = 0; kk < KCONV; ++kk) {
        int st = s - (KCONV - 1) + kk;
#pragma unroll
        for (int i = 0; i < BLKK; ++i)
            xm[kk][i] = (st >= 0)
                ? x_inner[((size_t)(b * SS + st)) * (2 * FF) + f0 + i] : 0.f;
    }

    float xa[BLKK], xv[BLKK];
#pragma unroll
    for (int i = 0; i < BLKK; ++i) {
        float acc = conv_b[f0 + i];
#pragma unroll
        for (int kk = 0; kk < KCONV; ++kk)
            acc += conv_w[kk * FF + f0 + i] * xm[kk][i];
        xa[i] = silu_f(acc);
        xv[i] = xm[KCONV - 1][i];   // x_mlstm at s
    }

    // 4x4 block-diagonal projections
    float qv[BLKK], kv[BLKK], vv[BLKK];
#pragma unroll
    for (int o = 0; o < BLKK; ++o) {
        float a0 = 0.f, a1 = 0.f, a2 = 0.f;
#pragma unroll
        for (int i = 0; i < BLKK; ++i) {
            a0 += q_w[(p * BLKK + o) * BLKK + i] * xa[i];
            a1 += k_w[(p * BLKK + o) * BLKK + i] * xa[i];
            a2 += v_w[(p * BLKK + o) * BLKK + i] * xv[i];
        }
        qv[o] = a0; kv[o] = a1; vv[o] = a2;
    }

    const size_t base = (size_t)bs * FF + f0;
#pragma unroll
    for (int o = 0; o < BLKK; ++o) {
        q16[base + o] = (_Float16)qv[o];
        k16[base + o] = (_Float16)kv[o];
        x_act[base + o] = xa[o];
        // transposed V: (b, h, dh, s)
        const int f = f0 + o;
        const int h = f >> 8;          // / DHH
        const int d = f & (DHH - 1);
        vT16[((size_t)(b * NHH + h) * DHH + d) * SS + s] = (_Float16)vv[o];
    }

    // gate logits: qkv (3F) . ig_w[:,h] / fg_w[:,h]
    __shared__ float red[8][256];
    float gp[8];
#pragma unroll
    for (int h = 0; h < NHH; ++h) {
        float gi = 0.f, gf = 0.f;
#pragma unroll
        for (int o = 0; o < BLKK; ++o) {
            const int r0 = f0 + o, r1 = FF + f0 + o, r2 = 2 * FF + f0 + o;
            gi += qv[o] * ig_w[r0 * NHH + h] + kv[o] * ig_w[r1 * NHH + h]
                + vv[o] * ig_w[r2 * NHH + h];
            gf += qv[o] * fg_w[r0 * NHH + h] + kv[o] * fg_w[r1 * NHH + h]
                + vv[o] * fg_w[r2 * NHH + h];
        }
        gp[h] = gi; gp[4 + h] = gf;
    }
#pragma unroll
    for (int g = 0; g < 8; ++g) red[g][p] = gp[g];
    __syncthreads();
    for (int off = 128; off > 0; off >>= 1) {
        if (p < off)
#pragma unroll
            for (int g = 0; g < 8; ++g) red[g][p] += red[g][p + off];
        __syncthreads();
    }
    if (p < NHH) {
        ig_out [((size_t)(b * NHH + p)) * SS + s] = red[p][0]     + ig_b[p];
        fgl_out[((size_t)(b * NHH + p)) * SS + s] = red[4 + p][0] + fg_b[p];
    }
}

// ---------------------------------------------------------------------------
// K2: per-(b,h) inclusive prefix sum of log-sigmoid(fg). grid=16, 1 wave32.
// ---------------------------------------------------------------------------
__global__ __launch_bounds__(32)
void k2_scan(const float* __restrict__ fgl, float* __restrict__ cum)
{
    const int bh = blockIdx.x;
    const int lane = threadIdx.x;
    const float* f = fgl + (size_t)bh * SS;
    float* c = cum + (size_t)bh * (SS + 1);
    if (lane == 0) c[0] = 0.f;
    float carry = 0.f;
    for (int s0 = 0; s0 < SS; s0 += 32) {
        float x = f[s0 + lane];
        float v = fminf(x, 0.f) - log1pf(__expf(-fabsf(x)));   // log-sigmoid
#pragma unroll
        for (int off = 1; off < 32; off <<= 1) {
            float n = __shfl_up(v, off);
            if (lane >= off) v += n;
        }
        v += carry;
        c[s0 + lane + 1] = v;
        carry = __shfl(v, 31);
    }
}

// ---------------------------------------------------------------------------
// K3: flash-mLSTM over 16-row tiles, 32-column steps. One wave32 per tile.
// Per step: 16x WMMA QK^T (two N-subtiles share each A load) + 16x WMMA P.V
// with a full 16x32 f16 P tile (no zero-padded K half).
// ---------------------------------------------------------------------------
__global__ __launch_bounds__(32)
void k3_mlstm(const _Float16* __restrict__ q16, const _Float16* __restrict__ k16,
              const _Float16* __restrict__ vT16,
              const float* __restrict__ ig, const float* __restrict__ cum,
              const float* __restrict__ ln_w, float* __restrict__ h_out)
{
    int id = blockIdx.x;
    const int it = id % TILES; id /= TILES;
    const int h  = id % NHH;   id /= NHH;
    const int b  = id;
    const int lane = threadIdx.x;
    const int row0 = it * 16;
    const int half = lane >> 4;       // which 16-lane group
    const int nn   = lane & 15;       // column within tile (C/D layout)
    const int kaddA = half ? 16 : 0;  // A-operand K base adj
    const int kaddB = half ? 16 : 0;  // B-operand K base adj

    __shared__ _Float16 qs[16 * DHH];   // Q tile, row-major 16 x 256
    __shared__ _Float16 ps[16 * 32];    // P tile f16, full 16 x 32

    {   // stage Q tile into LDS
        const _Float16* qg = q16 + ((size_t)(b * SS + row0)) * FF + h * DHH;
        for (int idx = lane; idx < 16 * DHH; idx += 32) {
            int r = idx >> 8, d = idx & 255;
            qs[r * DHH + d] = qg[(size_t)r * FF + d];
        }
    }
    __syncthreads();

    const float* cumbh = cum + (size_t)(b * NHH + h) * (SS + 1);
    const float* igbh  = ig  + (size_t)(b * NHH + h) * SS;

    float cumr[8];
#pragma unroll
    for (int r = 0; r < 8; ++r) cumr[r] = cumbh[row0 + r + 8 * half + 1];

    v8f acc[16];
#pragma unroll
    for (int t = 0; t < 16; ++t) acc[t] = (v8f)0.f;
    float mrun[8], lrun[8];
#pragma unroll
    for (int r = 0; r < 8; ++r) { mrun[r] = -__builtin_inff(); lrun[r] = 0.f; }

    const float inv_sqrt_dh = 0.0625f;  // 1/sqrt(256)
    const _Float16* vtg = vT16 + (size_t)(b * NHH + h) * DHH * SS;

    const int nsteps = (it >> 1) + 1;   // 32-column steps
    for (int jt = 0; jt < nsteps; ++jt) {
        const int col0 = jt * 32;
        const _Float16* kg = k16 + ((size_t)(b * SS + col0)) * FF + h * DHH;
        if (jt + 1 < nsteps) {  // prefetch next K/V step into GL2
            __builtin_prefetch(kg + (size_t)32 * FF + lane * 32, 0, 1);
            __builtin_prefetch(vtg + (size_t)(lane * 8) * SS + col0 + 32, 0, 1);
        }

        // ---- S = Q K^T over DH=256 in 8 K-chunks; two 16-col N-subtiles
        v8f sv0 = (v8f)0.f, sv1 = (v8f)0.f;
#pragma unroll
        for (int kc = 0; kc < DHH / 32; ++kc) {
            v16h a;                           // A: 16x32 f16, row = lane&15
#pragma unroll
            for (int i = 0; i < 16; ++i) {
                int kk = i + kaddA;
                a[i] = qs[nn * DHH + kc * 32 + kk];
            }
            v16h bb0, bb1;                    // B: K=dh chunk, N=col
#pragma unroll
            for (int i = 0; i < 16; ++i) {
                bb0[i] = kg[(size_t)nn * FF + kc * 32 + i + kaddB];
                bb1[i] = kg[(size_t)(nn + 16) * FF + kc * 32 + i + kaddB];
            }
            sv0 = __builtin_amdgcn_wmma_f32_16x16x32_f16(
                      false, a, false, bb0, (short)0, sv0, false, false);
            sv1 = __builtin_amdgcn_wmma_f32_16x16x32_f16(
                      false, a, false, bb1, (short)0, sv1, false, false);
        }

        // ---- gating + online stabilization over 32 columns
        const int c0 = col0 + nn;
        const int c1 = c0 + 16;
        const float cumc0 = cumbh[c0 + 1], cumc1 = cumbh[c1 + 1];
        const float igc0  = igbh[c0],      igc1  = igbh[c1];
        float pv0s[8], pv1s[8];
#pragma unroll
        for (int r = 0; r < 8; ++r) {
            const int row = row0 + r + 8 * half;
            float ld0 = (row >= c0) ? (cumr[r] - cumc0 + igc0) : -__builtin_inff();
            float ld1 = (row >= c1) ? (cumr[r] - cumc1 + igc1) : -__builtin_inff();
            float rm = fmaxf(ld0, ld1);
#pragma unroll
            for (int m = 1; m < 16; m <<= 1) rm = fmaxf(rm, __shfl_xor(rm, m));
            const float mn = fmaxf(mrun[r], rm);
            const float scale = (mrun[r] == -__builtin_inff())
                                ? 0.f : __expf(mrun[r] - mn);
#pragma unroll
            for (int t = 0; t < 16; ++t) acc[t][r] *= scale;
            const float d0 = (ld0 == -__builtin_inff()) ? 0.f : __expf(ld0 - mn);
            const float d1 = (ld1 == -__builtin_inff()) ? 0.f : __expf(ld1 - mn);
            const float p0 = sv0[r] * inv_sqrt_dh * d0;
            const float p1 = sv1[r] * inv_sqrt_dh * d1;
            pv0s[r] = p0; pv1s[r] = p1;
            float rs = p0 + p1;
#pragma unroll
            for (int m = 1; m < 16; m <<= 1) rs += __shfl_xor(rs, m);
            lrun[r] = lrun[r] * scale + rs;
            mrun[r] = mn;
        }

        // ---- P to LDS (full 16x32 f16)
#pragma unroll
        for (int r = 0; r < 8; ++r) {
            const int mrow = r + 8 * half;
            ps[mrow * 32 + nn]      = (_Float16)pv0s[r];
            ps[mrow * 32 + 16 + nn] = (_Float16)pv1s[r];
        }
        __syncthreads();

        // ---- acc += P . V  (16 N-tiles cover DH=256; K=32 fully used)
        v16h ap;
#pragma unroll
        for (int i = 0; i < 16; ++i) {
            int kk = i + kaddA;
            ap[i] = ps[nn * 32 + kk];
        }
#pragma unroll
        for (int t = 0; t < 16; ++t) {
            v16h bb;                          // B from transposed V: contiguous
#pragma unroll
            for (int i = 0; i < 16; ++i)
                bb[i] = vtg[(size_t)(t * 16 + nn) * SS + col0 + i + kaddB];
            acc[t] = __builtin_amdgcn_wmma_f32_16x16x32_f16(
                         false, ap, false, bb, (short)0, acc[t], false, false);
        }
        __syncthreads();
    }

    // ---- finalize: normalizer n, then per-row LayerNorm * ln_w
    float* hb = h_out + ((size_t)(b * SS + row0)) * FF + h * DHH;
#pragma unroll
    for (int r = 0; r < 8; ++r) {
        const float nr  = fmaxf(fabsf(lrun[r]), __expf(-mrun[r]));
        const float inv = 1.f / (nr + 1e-6f);
        float hv[16], sum = 0.f, sq = 0.f;
#pragma unroll
        for (int t = 0; t < 16; ++t) {
            float x = acc[t][r] * inv;
            hv[t] = x; sum += x; sq += x * x;
        }
#pragma unroll
        for (int m = 1; m < 16; m <<= 1) {
            sum += __shfl_xor(sum, m);
            sq  += __shfl_xor(sq,  m);
        }
        const float mu   = sum * (1.f / (float)DHH);
        const float var  = sq * (1.f / (float)DHH) - mu * mu;
        const float rstd = rsqrtf(var + 1e-5f);
        const int row = r + 8 * half;
#pragma unroll
        for (int t = 0; t < 16; ++t) {
            const float w = ln_w[h * DHH + t * 16 + nn];
            hb[(size_t)row * FF + t * 16 + nn] = (hv[t] - mu) * rstd * w;
        }
    }
}

// ---------------------------------------------------------------------------
// K4: out = (h_tilde + skip * x_act) * swish(z)
// ---------------------------------------------------------------------------
__global__ __launch_bounds__(256)
void k4_output(const float* __restrict__ x_inner, const float* __restrict__ skip,
               const float* __restrict__ h_tilde, const float* __restrict__ x_act,
               float* __restrict__ out)
{
    const size_t i  = (size_t)blockIdx.x * 256 + threadIdx.x;
    const int    f  = (int)(i & (FF - 1));
    const size_t bs = i >> 10;                        // FF = 1024
    const float  z  = x_inner[bs * (2 * FF) + FF + f];
    out[i] = (h_tilde[i] + skip[f] * x_act[i]) * silu_f(z);
}

// ---------------------------------------------------------------------------
extern "C" void kernel_launch(void* const* d_in, const int* in_sizes, int n_in,
                              void* d_out, int out_size, void* d_ws, size_t ws_size,
                              hipStream_t stream)
{
    const float* x_inner = (const float*)d_in[0];
    const float* conv_w  = (const float*)d_in[1];
    const float* conv_b  = (const float*)d_in[2];
    const float* q_w     = (const float*)d_in[3];
    const float* k_w     = (const float*)d_in[4];
    const float* v_w     = (const float*)d_in[5];
    const float* ig_w    = (const float*)d_in[6];
    const float* ig_b    = (const float*)d_in[7];
    const float* fg_w    = (const float*)d_in[8];
    const float* fg_b    = (const float*)d_in[9];
    const float* ln_w    = (const float*)d_in[10];
    const float* skip    = (const float*)d_in[11];
    float* out = (float*)d_out;

    const size_t nBSF = (size_t)BB * SS * FF;        // 8,388,608
    char* ws = (char*)d_ws;
    _Float16* q16   = (_Float16*)(ws);
    _Float16* k16   = (_Float16*)(ws + nBSF * 2);
    _Float16* vT16  = (_Float16*)(ws + nBSF * 4);
    float* x_act    = (float*)   (ws + nBSF * 6);
    float* h_tilde  = (float*)   (ws + nBSF * 6 + nBSF * 4);
    float* ig_buf   = (float*)   (ws + nBSF * 6 + nBSF * 8);
    float* fgl_buf  = (float*)   (ws + nBSF * 6 + nBSF * 8 + (size_t)BB * NHH * SS * 4);
    float* cum_buf  = (float*)   (ws + nBSF * 6 + nBSF * 8 + (size_t)BB * NHH * SS * 8);

    // K1: conv + swish + q/k/v + gate logits
    k1_qkv_gates<<<BB * SS, 256, 0, stream>>>(
        x_inner, conv_w, conv_b, q_w, k_w, v_w,
        ig_w, ig_b, fg_w, fg_b,
        q16, k16, vT16, x_act, ig_buf, fgl_buf);

    // K2: forget-gate log-sigmoid prefix sums
    k2_scan<<<BB * NHH, 32, 0, stream>>>(fgl_buf, cum_buf);

    // K3: flash-mLSTM (WMMA)
    k3_mlstm<<<BB * NHH * TILES, 32, 0, stream>>>(
        q16, k16, vT16, ig_buf, cum_buf, ln_w, h_tilde);

    // K4: skip + output gating
    k4_output<<<(int)(nBSF / 256), 256, 0, stream>>>(
        x_inner, skip, h_tilde, x_act, out);
}
